// MHA_33311766348262
// MI455X (gfx1250) — matrix operand order repaired and checked by reading, not compiled, not run on hardware
//
#include <hip/hip_runtime.h>
#include <hip/hip_bf16.h>

// ---------------------------------------------------------------------------
// MHA forward for MI455X (gfx1250): bf16 WMMA + TDM-staged GEMM & attention.
//   S=2048 seq, D=2048 model, H=16 heads, E=128 head dim, 3E=384.
// ---------------------------------------------------------------------------

typedef __bf16 bf16;
typedef __attribute__((ext_vector_type(16))) __bf16 v16bf;
typedef __attribute__((ext_vector_type(8)))  __bf16 v8bf;
typedef __attribute__((ext_vector_type(8)))  float  v8f;
typedef __attribute__((ext_vector_type(4)))  unsigned int v4u;
typedef __attribute__((ext_vector_type(8)))  int v8i;
typedef __attribute__((ext_vector_type(4)))  int v4i;

#define SQ   2048   // sequence length
#define DM   2048   // d_model
#define NH   16     // heads
#define EH   128    // head dim
#define N3   384    // 3*EH

static __device__ __forceinline__ v16bf cat8(v8bf lo, v8bf hi) {
  return __builtin_shufflevector(lo, hi, 0,1,2,3,4,5,6,7,8,9,10,11,12,13,14,15);
}

static __device__ __forceinline__ v8f wmma_bf16(v16bf a, v16bf b, v8f c) {
  return __builtin_amdgcn_wmma_f32_16x16x32_bf16(false, a, false, b,
                                                 (short)0, c, false, false);
}

// Byte offset of a __shared__ object within the LDS segment (generic -> AS3).
static __device__ __forceinline__ unsigned lds_offset(const void* p) {
  return (unsigned)(unsigned long long)
      (__attribute__((address_space(3))) const void*)p;
}

// TDM 2D tile load: global (tile_w x tile_h, 2-byte elems, row stride in
// elements) -> contiguous LDS.  D# per CDNA5 ISA:
//   group0: count=1 | lds_addr | global_addr | type=2
//   group1: data_size=2B, tensor_dim0/1 = tile dims, tile_dim0/1, dim0 stride
//   trailing groups: zero (2D tile, no iteration/gather, no multicast)
static __device__ __forceinline__ void tdm_load_2d(
    unsigned lds_byte_off, const void* gptr,
    unsigned tile_w, unsigned tile_h, unsigned long long row_stride_elems) {
  unsigned long long ga = (unsigned long long)(size_t)gptr;
  v4u g0;
  g0.x = 1u;                                              // count=1, user mode
  g0.y = lds_byte_off;                                    // lds_addr [63:32]
  g0.z = (unsigned)(ga & 0xFFFFFFFFu);                    // global_addr lo
  g0.w = (unsigned)((ga >> 32) & 0x01FFFFFFu) | (2u << 30); // addr hi | type=2
  v8i g1;
  g1[0] = (int)(1u << 16);                                // data_size = 2 bytes
  g1[1] = (int)((tile_w & 0xFFFFu) << 16);                // tensor_dim0[15:0]
  g1[2] = (int)(((tile_w >> 16) & 0xFFFFu) |              // tensor_dim0[31:16]
                ((tile_h & 0xFFFFu) << 16));              // tensor_dim1[15:0]
  g1[3] = (int)(((tile_h >> 16) & 0xFFFFu) |              // tensor_dim1[31:16]
                ((tile_w & 0xFFFFu) << 16));              // tile_dim0
  g1[4] = (int)(tile_h & 0xFFFFu);                        // tile_dim1 (dim2=0)
  g1[5] = (int)(unsigned)(row_stride_elems & 0xFFFFFFFFu);   // dim0_stride lo
  g1[6] = (int)(unsigned)((row_stride_elems >> 32) & 0xFFFFu);
  g1[7] = 0;
  v4i z4 = {0, 0, 0, 0};
  v8i z8 = {0, 0, 0, 0, 0, 0, 0, 0};
  // 6-arg form (clang-23 / therock-10.0 headers).
  __builtin_amdgcn_tensor_load_to_lds(g0, g1, z4, z4, z8, 0);
}

// ---------------------------------------------------------------------------
// Kernel 1: x fp32 -> bf16 (row-major [S][D], A-matrix feed)
// ---------------------------------------------------------------------------
__global__ void __launch_bounds__(256)
cvt_x_kernel(const float* __restrict__ x, bf16* __restrict__ xb) {
  int i = (blockIdx.x * 256 + threadIdx.x) * 4;
  if (i + 3 < SQ * DM) {
    float4 f = *(const float4*)(x + i);
    xb[i + 0] = (bf16)f.x;
    xb[i + 1] = (bf16)f.y;
    xb[i + 2] = (bf16)f.z;
    xb[i + 3] = (bf16)f.w;
  }
}

// ---------------------------------------------------------------------------
// Kernel 2: w fp32 [H][D][3E] -> bf16 transposed wT [H][3E][D]
// ---------------------------------------------------------------------------
__global__ void __launch_bounds__(256)
cvt_wT_kernel(const float* __restrict__ w, bf16* __restrict__ wT) {
  int i = blockIdx.x * 256 + threadIdx.x;       // over H*D*3E = 12.58M
  if (i < NH * DM * N3) {
    int e  = i % N3;
    int hd = i / N3;
    int d  = hd % DM;
    int h  = hd / DM;
    wT[(h * N3 + e) * DM + d] = (bf16)w[i];
  }
}

// ---------------------------------------------------------------------------
// Kernel 3: QKV projection.  One block = 8 waves sharing one 64-column weight
// strip; wave w owns M rows [mg*128 + w*16, +16).  Per 64-deep K step the TDM
// stages the 64x64 bf16 wT panel into double-buffered LDS (issued by wave 0,
// drained with s_wait_tensorcnt, published by barriers); all 8 waves read B
// fragments from LDS (8x cut in L2 weight traffic) and do 8 WMMAs per step.
// A fragments are two 16-B global loads per half-step (L2-resident x rows).
// ---------------------------------------------------------------------------
__global__ void __launch_bounds__(256)
qkv_gemm_kernel(const bf16* __restrict__ xb, const bf16* __restrict__ wT,
                bf16* __restrict__ qb, bf16* __restrict__ kb,
                bf16* __restrict__ vT) {
  __shared__ __attribute__((aligned(128))) bf16 bbuf[2][64 * 64]; // 2 x 8 KB

  const int lane = threadIdx.x & 31;
  const int wave = threadIdx.x >> 5;
  const int h   = blockIdx.x / 96;              // 96 blocks per head
  const int rem = blockIdx.x % 96;
  const int m0  = (rem / 6) * 128 + wave * 16;  // 16 M-groups of 128 rows
  const int n0  = (rem % 6) * 64;               // 6 strips of 64 columns
  const int hi = lane >> 4;
  const int lo = lane & 15;

  const unsigned boff0 = lds_offset(bbuf[0]);
  const unsigned boff1 = lds_offset(bbuf[1]);
  const bf16* arow   = xb + (m0 + lo) * DM + hi * 16;
  const bf16* wpanel = wT + (h * N3 + n0) * DM;

  v8f acc[4] = {};

  // TDM prologue: stage K panel k=0 into buffer 0.
  if (wave == 0) {
    tdm_load_2d(boff0, wpanel, 64, 64, DM);
    __builtin_amdgcn_s_wait_tensorcnt(0);
  }
  __syncthreads();

  for (int k = 0; k < DM; k += 64) {
    const int cur = (k >> 6) & 1;
    if (wave == 0 && k + 64 < DM)
      tdm_load_2d(cur ? boff0 : boff1, wpanel + k + 64, 64, 64, DM);

    v16bf a0 = cat8(*(const v8bf*)(arow + k),
                    *(const v8bf*)(arow + k + 8));
    v16bf a1 = cat8(*(const v8bf*)(arow + k + 32),
                    *(const v8bf*)(arow + k + 40));
    const bf16* bp = bbuf[cur];
#pragma unroll
    for (int j = 0; j < 4; ++j) {
      v16bf b = *(const v16bf*)(bp + (j * 16 + lo) * 64 + hi * 16);
      acc[j] = wmma_bf16(a0, b, acc[j]);
    }
#pragma unroll
    for (int j = 0; j < 4; ++j) {
      v16bf b = *(const v16bf*)(bp + (j * 16 + lo) * 64 + 32 + hi * 16);
      acc[j] = wmma_bf16(a1, b, acc[j]);
    }

    if (wave == 0) __builtin_amdgcn_s_wait_tensorcnt(0);
    __syncthreads();
  }

  const float scale = 0.08838834764831845f;     // 1/sqrt(128)
#pragma unroll
  for (int j = 0; j < 4; ++j) {
    int nc = n0 + j * 16 + lo;                  // output column (0..383)
#pragma unroll
    for (int r = 0; r < 8; ++r) {
      int m = m0 + r + 8 * hi;                  // output row
      float v = acc[j][r];
      if (nc < EH) {
        qb[(h * SQ + m) * EH + nc] = (bf16)(v * scale);
      } else if (nc < 2 * EH) {
        kb[(h * SQ + m) * EH + (nc - EH)] = (bf16)v;
      } else {
        vT[(h * EH + (nc - 2 * EH)) * SQ + m] = (bf16)v;
      }
    }
  }
}

// ---------------------------------------------------------------------------
// Kernel 4: flash attention.  One block = 4 waves = one head, 64 query rows.
// TDM double-buffers the shared 32x128 K tile and 128x32 V^T tile into LDS;
// all 4 waves consume them.  Per 32-column step each wave does 8 score WMMAs,
// online softmax, a C->A transpose of P through its private LDS slice, and
// 8 PV WMMAs.
// ---------------------------------------------------------------------------
__global__ void __launch_bounds__(128)
attn_kernel(const bf16* __restrict__ qb, const bf16* __restrict__ kb,
            const bf16* __restrict__ vT, float* __restrict__ out) {
  __shared__ __attribute__((aligned(128))) bf16 kbuf[2][32 * EH]; // 2 x 8 KB
  __shared__ __attribute__((aligned(128))) bf16 vbuf[2][EH * 32]; // 2 x 8 KB
  __shared__ __attribute__((aligned(128))) bf16 pstage[4][16 * 32]; // 4 x 1 KB

  const int lane = threadIdx.x & 31;
  const int wave = threadIdx.x >> 5;
  const int h  = blockIdx.x >> 5;               // 32 blocks per head
  const int m0 = (blockIdx.x & 31) * 64 + wave * 16;
  const int hi = lane >> 4;
  const int lo = lane & 15;
  bf16* myp = pstage[wave];

  const unsigned koff0 = lds_offset(kbuf[0]);
  const unsigned koff1 = lds_offset(kbuf[1]);
  const unsigned voff0 = lds_offset(vbuf[0]);
  const unsigned voff1 = lds_offset(vbuf[1]);

  // Preload Q A-fragments (4 K-slabs of 32 over E=128); 1/sqrt(E) pre-folded.
  v16bf qf[4];
  {
    const bf16* qrow = qb + (h * SQ + m0 + lo) * EH + hi * 16;
#pragma unroll
    for (int i = 0; i < 4; ++i)
      qf[i] = cat8(*(const v8bf*)(qrow + i * 32),
                   *(const v8bf*)(qrow + i * 32 + 8));
  }

  v8f acc[8] = {};                              // 16 x 128 output tile
  float mrow[8], lrow[8];
#pragma unroll
  for (int r = 0; r < 8; ++r) { mrow[r] = -__builtin_inff(); lrow[r] = 0.f; }

  // TDM prologue: stage tile t0=0 into buffer 0.
  if (wave == 0) {
    tdm_load_2d(koff0, kb + (h * SQ) * EH, EH, 32, EH);
    tdm_load_2d(voff0, vT + (h * EH) * SQ, 32, EH, SQ);
    __builtin_amdgcn_s_wait_tensorcnt(0);
  }
  __syncthreads();

  for (int t0 = 0; t0 < SQ; t0 += 32) {
    const int cur = (t0 >> 5) & 1;
    // Issue TDM for the next tile into the alternate buffer.
    if (wave == 0 && t0 + 32 < SQ) {
      tdm_load_2d(cur ? koff0 : koff1, kb + (h * SQ + t0 + 32) * EH, EH, 32, EH);
      tdm_load_2d(cur ? voff0 : voff1, vT + (h * EH) * SQ + t0 + 32, 32, EH, SQ);
    }
    const bf16* kt = kbuf[cur];
    const bf16* vt = vbuf[cur];

    // --- scores: two 16x16 tiles (cols t0..t0+15 and t0+16..t0+31) ---
    v8f c0 = {}, c1 = {};
#pragma unroll
    for (int es = 0; es < EH; es += 32) {
      v16bf kf0 = *(const v16bf*)(kt + lo * EH + hi * 16 + es);
      v16bf kf1 = *(const v16bf*)(kt + (16 + lo) * EH + hi * 16 + es);
      c0 = wmma_bf16(qf[es >> 5], kf0, c0);
      c1 = wmma_bf16(qf[es >> 5], kf1, c1);
    }

    // --- online softmax; emit P (bf16) to private LDS slice [16][32] ---
#pragma unroll
    for (int r = 0; r < 8; ++r) {
      float v = fmaxf(c0[r], c1[r]);
#pragma unroll
      for (int s = 1; s < 16; s <<= 1) v = fmaxf(v, __shfl_xor(v, s, 16));
      float mn    = fmaxf(mrow[r], v);
      float alpha = __expf(mrow[r] - mn);
      float p0 = __expf(c0[r] - mn);
      float p1 = __expf(c1[r] - mn);
      float rs = p0 + p1;
#pragma unroll
      for (int s = 1; s < 16; s <<= 1) rs += __shfl_xor(rs, s, 16);
      lrow[r] = lrow[r] * alpha + rs;
      mrow[r] = mn;
#pragma unroll
      for (int j = 0; j < 8; ++j) acc[j][r] *= alpha;
      int m = r + 8 * hi;                       // P row
      myp[m * 32 + lo]      = (bf16)p0;
      myp[m * 32 + 16 + lo] = (bf16)p1;
    }

    // Cross-lane LDS dependency inside the wave -> explicit dscnt drain.
    __builtin_amdgcn_wave_barrier();
    asm volatile("s_wait_dscnt 0x0" ::: "memory");
    __builtin_amdgcn_wave_barrier();

    // --- reload P as A-fragment (lane = row) and do PV from the LDS V tile ---
    const bf16* pr = myp + lo * 32 + hi * 16;
    v16bf pf = cat8(*(const v8bf*)pr, *(const v8bf*)(pr + 8));
#pragma unroll
    for (int j = 0; j < 8; ++j) {
      v16bf vf = *(const v16bf*)(vt + (j * 16 + lo) * 32 + hi * 16);
      acc[j] = wmma_bf16(pf, vf, acc[j]);
    }

    // Publish the next tile: wave 0 drains TENSORcnt, everyone barriers.
    if (wave == 0) __builtin_amdgcn_s_wait_tensorcnt(0);
    __syncthreads();
  }

  // --- normalize and write fp32 output [s][h*E + e] ---
#pragma unroll
  for (int r = 0; r < 8; ++r) {
    float inv = 1.0f / lrow[r];
    int m = m0 + r + 8 * hi;
#pragma unroll
    for (int j = 0; j < 8; ++j)
      out[m * DM + h * EH + j * 16 + lo] = acc[j][r] * inv;
  }
}

// ---------------------------------------------------------------------------
// Launch: convert -> project -> attend.  Workspace layout (bytes):
//   xb  bf16 [S][D]        8 MB
//   wT  bf16 [H][3E][D]   24 MB
//   qb  bf16 [H][S][E]     8 MB
//   kb  bf16 [H][S][E]     8 MB
//   vT  bf16 [H][E][S]     8 MB   (total 56 MB)
// ---------------------------------------------------------------------------
extern "C" void kernel_launch(void* const* d_in, const int* in_sizes, int n_in,
                              void* d_out, int out_size, void* d_ws, size_t ws_size,
                              hipStream_t stream) {
  (void)in_sizes; (void)n_in; (void)out_size; (void)ws_size;
  const float* x = (const float*)d_in[0];
  const float* w = (const float*)d_in[1];
  float* out = (float*)d_out;

  char* ws = (char*)d_ws;
  bf16* xb = (bf16*)ws;
  bf16* wT = (bf16*)(ws + (size_t)SQ * DM * 2);
  bf16* qb = (bf16*)(ws + (size_t)SQ * DM * 2 + (size_t)NH * N3 * DM * 2);
  bf16* kb = qb + (size_t)NH * SQ * EH;
  bf16* vT = kb + (size_t)NH * SQ * EH;

  cvt_x_kernel <<<(SQ * DM / 4 + 255) / 256, 256, 0, stream>>>(x, xb);
  cvt_wT_kernel<<<(NH * DM * N3 + 255) / 256, 256, 0, stream>>>(w, wT);
  qkv_gemm_kernel<<<NH * 96, 256, 0, stream>>>(xb, wT, qb, kb, vT);
  attn_kernel   <<<NH * 32, 128, 0, stream>>>(qb, kb, vT, out);
}
